// DependentLatentModel_73426760892519
// MI455X (gfx1250) — matrix operand order repaired and checked by reading, not compiled, not run on hardware
//
#include <hip/hip_runtime.h>
#include <hip/hip_bf16.h>

typedef __bf16  bf16_t;
typedef __bf16  v16bf __attribute__((ext_vector_type(16)));
typedef float   v8f   __attribute__((ext_vector_type(8)));
typedef float   v4f   __attribute__((ext_vector_type(4)));
typedef unsigned int u32x4 __attribute__((ext_vector_type(4)));
typedef int          i32x8 __attribute__((ext_vector_type(8)));
typedef int          i32x4 __attribute__((ext_vector_type(4)));

#define ENC    768
#define ZR     30
#define T_LEN  512
#define B_SZ   1024
#define NCOL   128            // 120 LSTM gate cols + a(120) + b(121) + pad
#define MROWS  64             // M-chunk rows staged in LDS per block
#define KUMA_L (-0.1f)
#define KUMA_R (1.1f)
#define EPSF   1e-6f

// ---------------------------------------------------------------------------
// Kernel 0: pack [768 x 122] weight columns (W_ih rows 0..119, Wa, Wb) into a
// transposed bf16 matrix wt[128][768] so each WMMA B-fragment lane reads 16
// consecutive bf16 (one 32-byte load).
// ---------------------------------------------------------------------------
__global__ __launch_bounds__(256) void pack_weights_kernel(
    const float* __restrict__ W_ih, const float* __restrict__ Wa,
    const float* __restrict__ Wb, __hip_bfloat16* __restrict__ wt)
{
  int idx = blockIdx.x * 256 + threadIdx.x;     // 0 .. 128*768-1 (exact)
  int n = idx / ENC;
  int k = idx - n * ENC;
  float v = 0.0f;
  if (n < 120)       v = W_ih[n * (ENC + 1) + k];
  else if (n == 120) v = Wa[k];
  else if (n == 121) v = Wb[k];
  wt[idx] = __float2bfloat16(v);
}

// ---------------------------------------------------------------------------
// Kernel 1: gates_h[B*T, 128] = h[B*T, 768] @ wt.T
// Block = 256 threads = 8 waves; block owns a 64-row M-chunk staged into LDS
// by the Tensor Data Mover (async DMA, TENSORcnt). Wave w computes M-subtile
// (w>>1) x N-half ((w&1)*64): each A fragment (from LDS) feeds 4 WMMAs.
// 24 k-steps x 4 N-tiles = 96 v_wmma_f32_16x16x32_bf16 per wave.
// ---------------------------------------------------------------------------
__global__ __launch_bounds__(256) void gemm_gates_kernel(
    const float* __restrict__ h, const __hip_bfloat16* __restrict__ wtp,
    float* __restrict__ gates)
{
  extern __shared__ float As[];                 // [MROWS][ENC] f32, 192 KB
  const bf16_t* __restrict__ wt = reinterpret_cast<const bf16_t*>(wtp);
  const int lane = threadIdx.x & 31;
  const int wv   = threadIdx.x >> 5;            // 0..7
  const int half = lane >> 4;
  const int l16  = lane & 15;
  const int ms   = wv >> 1;                     // M-subtile 0..3
  const int nb   = (wv & 1) * 64;               // N-half base col
  const long m0  = (long)blockIdx.x * MROWS;    // chunk base row

#if defined(__gfx1250__) && __has_builtin(__builtin_amdgcn_tensor_load_to_lds)
  // ---- Stage A chunk via Tensor Data Mover (wave 0 issues one descriptor) --
  if (wv == 0) {
    const unsigned long long ga =
        (unsigned long long)(__SIZE_TYPE__)(h + m0 * ENC);
    const unsigned int lds_off = (unsigned int)__builtin_amdgcn_groupstaticsize();
    // D# group 0: count=1 | lds_addr | global_addr[56:0] | type=2
    u32x4 g0;
    g0[0] = 1u;                                  // count=1, user mode
    g0[1] = lds_off;                             // LDS byte address
    g0[2] = (unsigned int)(ga & 0xFFFFFFFFu);    // global_addr[31:0]
    g0[3] = (unsigned int)((ga >> 32) & 0x01FFFFFFu) | (2u << 30); // [56:32]|type=2
    // D# group 1: data_size=4B, 2D tile 768 x 64, tensor dim0=768 stride=768
    i32x8 g1;
    g1[0] = (int)(2u << 16);                     // wg_mask=0, data_size=2 (4B)
    g1[1] = (int)(((unsigned)ENC & 0xFFFFu) << 16);        // tensor_dim0 lo
    g1[2] = (int)((((unsigned)ENC >> 16) & 0xFFFFu)        // tensor_dim0 hi
                  | (((unsigned)MROWS & 0xFFFFu) << 16));  // tensor_dim1 lo
    g1[3] = (int)((((unsigned)MROWS >> 16) & 0xFFFFu)      // tensor_dim1 hi
                  | (((unsigned)ENC & 0xFFFFu) << 16));    // tile_dim0
    g1[4] = (int)((unsigned)MROWS & 0xFFFFu);              // tile_dim1 (tile_dim2=0)
    g1[5] = (int)(unsigned)ENC;                            // tensor_dim0_stride[31:0]
    g1[6] = 0;                                             // stride hi, dim1_stride lo
    g1[7] = 0;
    i32x4 g2 = {0, 0, 0, 0};
    i32x4 g3 = {0, 0, 0, 0};
    i32x8 g4 = {0, 0, 0, 0, 0, 0, 0, 0};
    __builtin_amdgcn_tensor_load_to_lds(g0, g1, g2, g3, g4, 0);
    __builtin_amdgcn_s_wait_tensorcnt(0);
  }
  __syncthreads();
#else
  // ---- Fallback: cooperative vector copy h-chunk -> LDS -------------------
  {
    const v4f* src = (const v4f*)(h + m0 * ENC);
    v4f* dst = (v4f*)As;
    for (int i = threadIdx.x; i < MROWS * ENC / 4; i += 256) dst[i] = src[i];
  }
  __syncthreads();
#endif

  // B column pointers for this wave's 4 N-tiles (hot in L2: wt is 192 KB)
  const bf16_t* __restrict__ brow0 = wt + (nb +  0 + l16) * ENC;
  const bf16_t* __restrict__ brow1 = wt + (nb + 16 + l16) * ENC;
  const bf16_t* __restrict__ brow2 = wt + (nb + 32 + l16) * ENC;
  const bf16_t* __restrict__ brow3 = wt + (nb + 48 + l16) * ENC;

  v8f acc0 = {0.f,0.f,0.f,0.f,0.f,0.f,0.f,0.f};
  v8f acc1 = acc0, acc2 = acc0, acc3 = acc0;

  const float* __restrict__ arow = As + (ms * 16 + l16) * ENC;

  for (int k0 = 0; k0 < ENC; k0 += 32) {
    // A fragment (16x32 bf16) from LDS: K = ka+[0..16)
    const int ka = k0 + 16 * half;
    v4f a0 = *(const v4f*)(arow + ka);
    v4f a1 = *(const v4f*)(arow + ka + 4);
    v4f a2 = *(const v4f*)(arow + ka + 8);
    v4f a3 = *(const v4f*)(arow + ka + 12);
    v16bf A;
#pragma unroll
    for (int i = 0; i < 4; ++i) A[i]      = (bf16_t)a0[i];
#pragma unroll
    for (int i = 0; i < 4; ++i) A[4 + i]  = (bf16_t)a1[i];
#pragma unroll
    for (int i = 0; i < 4; ++i) A[8 + i]  = (bf16_t)a2[i];
#pragma unroll
    for (int i = 0; i < 4; ++i) A[12 + i] = (bf16_t)a3[i];

    const int kb = k0 + 16 * half;
    v16bf B0 = *(const v16bf*)(brow0 + kb);
    v16bf B1 = *(const v16bf*)(brow1 + kb);
    v16bf B2 = *(const v16bf*)(brow2 + kb);
    v16bf B3 = *(const v16bf*)(brow3 + kb);

    acc0 = __builtin_amdgcn_wmma_f32_16x16x32_bf16(false, A, false, B0, (short)0, acc0, false, false);
    acc1 = __builtin_amdgcn_wmma_f32_16x16x32_bf16(false, A, false, B1, (short)0, acc1, false, false);
    acc2 = __builtin_amdgcn_wmma_f32_16x16x32_bf16(false, A, false, B2, (short)0, acc2, false, false);
    acc3 = __builtin_amdgcn_wmma_f32_16x16x32_bf16(false, A, false, B3, (short)0, acc3, false, false);
  }

  // D layout: VGPR r -> row (ms*16 + 8*half + r), col (nb + nt*16 + l16)
  float* __restrict__ gout =
      gates + (m0 + ms * 16 + 8 * half) * NCOL + nb + l16;
#pragma unroll
  for (int r = 0; r < 8; ++r) {
    gout[(long)r * NCOL +  0] = acc0[r];
    gout[(long)r * NCOL + 16] = acc1[r];
    gout[(long)r * NCOL + 32] = acc2[r];
    gout[(long)r * NCOL + 48] = acc3[r];
  }
}

// ---------------------------------------------------------------------------
// Kernel 2: sequential HardKuma + LSTM scan. One wave32 per batch element.
// Lane l (<30) owns gate rows {l, 30+l, 60+l, 90+l}; hs replicated per lane.
// ---------------------------------------------------------------------------
__device__ __forceinline__ float softplus_f(float x) {
  return (x > 20.0f) ? x : __logf(1.0f + __expf(x));
}
__device__ __forceinline__ float sigmoid_f(float x) {
  return 1.0f / (1.0f + __expf(-x));
}
__device__ __forceinline__ float tanh_f(float x) {
  x = fminf(fmaxf(x, -15.0f), 15.0f);
  float e = __expf(2.0f * x);
  return (e - 1.0f) / (e + 1.0f);
}

__global__ __launch_bounds__(256) void hardkuma_scan_kernel(
    const unsigned char* __restrict__ mask, const float* __restrict__ u,
    const float* __restrict__ Wa, const float* __restrict__ ba,
    const float* __restrict__ Wb, const float* __restrict__ bb,
    const float* __restrict__ W_ih, const float* __restrict__ b_ih,
    const float* __restrict__ W_hh, const float* __restrict__ b_hh,
    const float* __restrict__ gates_h, float* __restrict__ out)
{
  const int lane = threadIdx.x & 31;
  const int b    = blockIdx.x * 8 + (threadIdx.x >> 5);
  const int gl   = (lane < ZR) ? lane : 0;      // gate row owned by this lane

  float whh[4][ZR], wz[4], bias[4];
#pragma unroll
  for (int q = 0; q < 4; ++q) {
    const int g = gl + ZR * q;
#pragma unroll
    for (int j = 0; j < ZR; ++j) whh[q][j] = W_hh[g * ZR + j];
    wz[q]   = W_ih[g * (ENC + 1) + ENC];        // z column of W_ih
    bias[q] = b_ih[g] + b_hh[g];
  }
  float waz[ZR], wbz[ZR];
#pragma unroll
  for (int j = 0; j < ZR; ++j) { waz[j] = Wa[ENC + j]; wbz[j] = Wb[ENC + j]; }
  const float bav = ba[0], bbv = bb[0];

  float hsv[ZR];
#pragma unroll
  for (int j = 0; j < ZR; ++j) hsv[j] = 0.0f;
  float cs = 0.0f;

  for (int t = 0; t < T_LEN; ++t) {
    const long bt = (long)b * T_LEN + t;
    const float* __restrict__ gh = gates_h + bt * NCOL;

    float arec = 0.0f, brec = 0.0f;
#pragma unroll
    for (int j = 0; j < ZR; ++j) {
      arec = fmaf(hsv[j], waz[j], arec);
      brec = fmaf(hsv[j], wbz[j], brec);
    }
    float av = fminf(fmaxf(softplus_f(gh[120] + arec + bav), EPSF), 100.0f);
    float bv = fminf(fmaxf(softplus_f(gh[121] + brec + bbv), EPSF), 100.0f);

    float uu = fminf(fmaxf(u[bt], EPSF), 1.0f - EPSF);
    float p  = __expf(__logf(1.0f - uu) / bv);                  // (1-u)^(1/b)
    float kk = __expf(__logf(fmaxf(1.0f - p, 1e-30f)) / av);    // (...)^(1/a)
    float z  = fminf(fmaxf(KUMA_L + (KUMA_R - KUMA_L) * kk, 0.0f), 1.0f);

    float gv[4];
#pragma unroll
    for (int q = 0; q < 4; ++q) {
      float s = gh[gl + ZR * q] + bias[q] + z * wz[q];
#pragma unroll
      for (int j = 0; j < ZR; ++j) s = fmaf(hsv[j], whh[q][j], s);
      gv[q] = s;
    }
    float ig = sigmoid_f(gv[0]);
    float fg = sigmoid_f(gv[1]);
    float gg = tanh_f(gv[2]);
    float og = sigmoid_f(gv[3]);
    cs = fg * cs + ig * gg;
    float hnew = og * tanh_f(cs);

#pragma unroll
    for (int j = 0; j < ZR; ++j) hsv[j] = __shfl(hnew, j, 32);

    if (lane == 0) out[bt] = mask[bt] ? z : 0.0f;
  }
}

// ---------------------------------------------------------------------------
extern "C" void kernel_launch(void* const* d_in, const int* in_sizes, int n_in,
                              void* d_out, int out_size, void* d_ws, size_t ws_size,
                              hipStream_t stream) {
  const float*         h     = (const float*)d_in[0];
  const unsigned char* mask  = (const unsigned char*)d_in[1];
  const float*         u     = (const float*)d_in[2];
  const float*         Wa    = (const float*)d_in[3];
  const float*         ba    = (const float*)d_in[4];
  const float*         Wb    = (const float*)d_in[5];
  const float*         bb    = (const float*)d_in[6];
  const float*         W_ih  = (const float*)d_in[7];
  const float*         b_ih  = (const float*)d_in[8];
  const float*         W_hh  = (const float*)d_in[9];
  const float*         b_hh  = (const float*)d_in[10];
  float*               out   = (float*)d_out;

  const size_t gates_bytes = (size_t)B_SZ * T_LEN * NCOL * sizeof(float); // 268.4 MB
  float*           ws_gates = (float*)d_ws;
  __hip_bfloat16*  ws_wt    = (__hip_bfloat16*)((char*)d_ws + gates_bytes);

  // 0) pack weights -> bf16 [128][768]
  pack_weights_kernel<<<(NCOL * ENC) / 256, 256, 0, stream>>>(W_ih, Wa, Wb, ws_wt);

  // 1) hoisted GEMM: [524288, 768] x [768, 128] -> gates_h (f32 accum)
  //    64-row M-chunk per block staged in 192 KB LDS via TDM.
  const size_t lds_bytes = (size_t)MROWS * ENC * sizeof(float);
  gemm_gates_kernel<<<(B_SZ * T_LEN) / MROWS, 256, lds_bytes, stream>>>(
      h, ws_wt, ws_gates);

  // 2) sequential scan: 1024 waves, 8 waves per block
  hardkuma_scan_kernel<<<B_SZ / 8, 256, 0, stream>>>(
      mask, u, Wa, ba, Wb, bb, W_ih, b_ih, W_hh, b_hh, ws_gates, out);
}
